// TrueeidosFFN_69741678952860
// MI455X (gfx1250) — compile-verified, matrix operand order checked by reading, not run on hardware
//
#include <hip/hip_runtime.h>
#include <hip/hip_bf16.h>
#include <math.h>
#include <stdint.h>

typedef __bf16 bf16_t;
typedef __attribute__((ext_vector_type(16))) __bf16 v16bf;
typedef __attribute__((ext_vector_type(8)))  __bf16 v8bf;
typedef __attribute__((ext_vector_type(8)))  float  v8f;
typedef __attribute__((ext_vector_type(4)))  unsigned int v4u;
typedef __attribute__((ext_vector_type(8)))  int v8i;
typedef __attribute__((ext_vector_type(4)))  int v4i;

#define DD 1024
#define FF 4096
#define PP 9
#define NROWS 8192   // B*S = 4*2048
#define KROT 4
#define BASEC 7.0f
#define EPSC 1e-6f

#ifndef USE_TDM
#define USE_TDM 1
#endif

// ---------------------------------------------------------------------------
// Fragment loader for 16-bit WMMA operands on gfx1250 (wave32).
// A (16x32, MxK) per ISA: lanes 0-15 hold K={0..7,16..23}, lanes 16-31 hold
// K={8..15,24..31} for row M=lane%16, packed as bf16 pairs. B from a
// K-contiguous (transposed) weight matrix has the mirrored layout with
// lane -> N column. Both reduce to two 16-byte vector loads per lane.
// ---------------------------------------------------------------------------
__device__ __forceinline__ v16bf frag16(const bf16_t* base, int stride, int row,
                                        int k0, int lane) {
  const int khalf = (lane >> 4) << 3;          // 0 or 8
  const bf16_t* q = base + (size_t)row * stride + k0 + khalf;
  v8bf lo = *(const v8bf*)q;                   // K  = k0 + khalf      .. +7
  v8bf hi = *(const v8bf*)(q + 16);            // K  = k0 + khalf + 16 .. +23
  v16bf r;
#pragma unroll
  for (int i = 0; i < 8; ++i) { r[i] = lo[i]; r[i + 8] = hi[i]; }
  return r;
}

__device__ __forceinline__ v8f wmma_bf16(v16bf a, v16bf b, v8f c) {
  // D(f32 16x16) = A(bf16 16x32) * B(bf16 32x16) + C
  return __builtin_amdgcn_wmma_f32_16x16x32_bf16(false, a, false, b,
                                                 (short)0, c, false, false);
}

// ---------------------------------------------------------------------------
// Stage 0: f32 -> bf16 cast (x activations)
// ---------------------------------------------------------------------------
__global__ void k_convert_bf16(const float* __restrict__ src,
                               bf16_t* __restrict__ dst, size_t n) {
  for (size_t i = (size_t)blockIdx.x * blockDim.x + threadIdx.x; i < n;
       i += (size_t)gridDim.x * blockDim.x)
    dst[i] = (bf16_t)src[i];
}

// ---------------------------------------------------------------------------
// Stage 1: convert + transpose weights to bf16, K-contiguous.
// src: [nmat][R][C] f32  ->  dst: [nmat][C][R] bf16
// ---------------------------------------------------------------------------
__global__ void k_transpose_to_bf16(const float* __restrict__ src,
                                    bf16_t* __restrict__ dst,
                                    int R, int C, int nmat) {
  const size_t per = (size_t)R * C;
  const size_t total = per * (size_t)nmat;
  for (size_t i = (size_t)blockIdx.x * blockDim.x + threadIdx.x; i < total;
       i += (size_t)gridDim.x * blockDim.x) {
    size_t m = i / per;
    size_t rem = i - m * per;
    int r = (int)(rem / C);
    int c = (int)(rem - (size_t)r * C);
    dst[m * per + (size_t)c * R + r] = (bf16_t)src[i];
  }
}

// ---------------------------------------------------------------------------
// Stage 2: fused per-path FFN. One block = 16 rows x one path, 8 waves.
// Sweep A: h1 = rot(x@W1 + b1), accumulate per-row sum of squares only.
// Sweep B: recompute h1 chunk (16x128) -> wrap/scale -> bf16 in LDS ->
//          GEMM2 accumulation in registers (each wave owns 16x128 of D).
// Epilogue: rot, RMS-wrap 2, path_bias, context*tanh, store f[p] (f32).
// x tile staged into LDS by the Tensor Data Mover (TDM).
// ---------------------------------------------------------------------------
__global__ __launch_bounds__(256)
void k_path(const bf16_t* __restrict__ xb,    // [NROWS][D] bf16
            const bf16_t* __restrict__ w1t,   // [P][F][D] bf16 (W1 transposed)
            const bf16_t* __restrict__ w2t,   // [P][D][F] bf16 (W2 transposed)
            const float* __restrict__ b1, const float* __restrict__ th1,
            const float* __restrict__ g1, const float* __restrict__ be1,
            const float* __restrict__ b2, const float* __restrict__ th2,
            const float* __restrict__ g2, const float* __restrict__ be2,
            const float* __restrict__ pbias, const float* __restrict__ ctx,
            float* __restrict__ fbuf)         // [P][NROWS][D] f32
{
  __shared__ __align__(16) bf16_t s_x[16 * DD];     // 32 KB: x row tile, bf16
  __shared__ __align__(16) bf16_t s_h1[16 * 128];   // 4 KB: wrapped h1 chunk
  __shared__ float s_sum[16];
  __shared__ float s_inv[16];

  const int p    = blockIdx.x / (NROWS / 16);
  const int tile = blockIdx.x % (NROWS / 16);
  const int r0   = tile * 16;
  const int tid  = threadIdx.x;
  const int lane = tid & 31;
  const int wave = tid >> 5;            // 0..7
  const int nn   = lane & 15;           // column within a 16-wide tile
  const int rbase = (lane >> 4) << 3;   // C layout: rows j (+0) or j+8

  const bf16_t* w1p = w1t + (size_t)p * FF * DD;   // row n (of F), stride D
  const bf16_t* w2p = w2t + (size_t)p * DD * FF;   // row d (of D), stride F

  if (tid < 16) s_sum[tid] = 0.0f;

#if USE_TDM
  // ---- stage x tile via Tensor Data Mover: one contiguous 32KB 1D tile ----
  if (wave == 0) {
    typedef __attribute__((address_space(3))) bf16_t* lds_bf16_p;
    const unsigned long long ga =
        (unsigned long long)(uintptr_t)(xb + (size_t)r0 * DD);  // global byte addr
    const unsigned int lds_base = (unsigned int)(uintptr_t)(lds_bf16_p)s_x;

    v4u g0 = {};
    g0[0] = 1u;                                        // count=1 (valid D#)
    g0[1] = lds_base;                                  // lds_addr
    g0[2] = (unsigned int)(ga & 0xffffffffu);          // global_addr[31:0]
    g0[3] = (unsigned int)((ga >> 32) & 0x01ffffffu)   // global_addr[56:32]
          | (2u << 30);                                // type = 2 ("image")

    const unsigned int td0 = (unsigned int)NROWS * DD; // tensor_dim0 (elements)
    v8i g1 = {};
    g1[0] = (int)(1u << 16);                           // data_size=1 -> 2 bytes
    g1[1] = (int)((td0 & 0xffffu) << 16);              // tensor_dim0[15:0] @ [63:48]
    g1[2] = (int)((td0 >> 16) | (1u << 16));           // tensor_dim0[31:16]; tensor_dim1=1
    g1[3] = (int)((16u * DD) << 16);                   // tile_dim0 = 16384 @ [127:112]
    g1[4] = 0;                                         // tile_dim1=0, tile_dim2=0
    g1[5] = (int)td0;                                  // tensor_dim0_stride[31:0]
    g1[6] = 0;
    g1[7] = 0;

    v4i g2 = {};
    v4i g3 = {};
#if defined(__clang_major__) && __clang_major__ >= 23
    v8i g2b = {};
    __builtin_amdgcn_tensor_load_to_lds(g0, g1, g2, g3, g2b, 0);
#else
    __builtin_amdgcn_tensor_load_to_lds(g0, g1, g2, g3, 0);
#endif
    __builtin_amdgcn_s_wait_tensorcnt(0);
  }
#else
  for (int i = tid; i < 16 * DD; i += 256)
    s_x[i] = xb[(size_t)r0 * DD + i];
#endif
  __syncthreads();

  // ================= sweep A: sum of squares of h1 =================
  float ps[8];
#pragma unroll
  for (int j = 0; j < 8; ++j) ps[j] = 0.0f;

  for (int ch = 0; ch < FF / 128; ++ch) {
    const int n = ch * 128 + wave * 16 + nn;       // column of F this lane owns
    if (n + 128 < FF) __builtin_prefetch(w1p + (size_t)(n + 128) * DD, 0, 1);
    v8f acc = {};
    for (int k0 = 0; k0 < DD; k0 += 32) {
      v16bf a = frag16(s_x, DD, nn, k0, lane);
      v16bf b = frag16(w1p, DD, n, k0, lane);
      acc = wmma_bf16(a, b, acc);
    }
    const float bias = b1[p * FF + n];
    const bool rot = (ch == 0) && (wave == 0);     // columns 0..7 of F
    float cc = 1.0f, ss = 0.0f;
    if (rot && nn < 8) { float th = th1[p * KROT + (nn >> 1)]; cc = cosf(th); ss = sinf(th); }
#pragma unroll
    for (int j = 0; j < 8; ++j) {
      float v = acc[j] + bias;
      if (rot) {
        float part = __shfl_xor(v, 1, 32);
        if (nn < 8) v = (nn & 1) ? (cc * v + ss * part) : (cc * v - ss * part);
      }
      ps[j] += v * v;
    }
  }
#pragma unroll
  for (int j = 0; j < 8; ++j) atomicAdd(&s_sum[rbase + j], ps[j]);
  __syncthreads();
  if (tid < 16) s_inv[tid] = rsqrtf(s_sum[tid] * (1.0f / FF) + EPSC);
  __syncthreads();

  // ============ sweep B: recompute h1 chunks, feed GEMM2 ============
  v8f c2[8];
  {
    v8f z = {};
#pragma unroll
    for (int t = 0; t < 8; ++t) c2[t] = z;
  }

  for (int ch = 0; ch < FF / 128; ++ch) {
    const int n = ch * 128 + wave * 16 + nn;
    v8f acc = {};
    for (int k0 = 0; k0 < DD; k0 += 32) {
      v16bf a = frag16(s_x, DD, nn, k0, lane);
      v16bf b = frag16(w1p, DD, n, k0, lane);
      acc = wmma_bf16(a, b, acc);
    }
    const float bias = b1[p * FF + n];
    const bool rot = (ch == 0) && (wave == 0);
    float cc = 1.0f, ss = 0.0f;
    if (rot && nn < 8) { float th = th1[p * KROT + (nn >> 1)]; cc = cosf(th); ss = sinf(th); }
    const float g  = g1[p * FF + n];
    const float be = be1[p * FF + n];
#pragma unroll
    for (int j = 0; j < 8; ++j) {
      float v = acc[j] + bias;
      if (rot) {
        float part = __shfl_xor(v, 1, 32);
        if (nn < 8) v = (nn & 1) ? (cc * v + ss * part) : (cc * v - ss * part);
      }
      const int r = rbase + j;
      float xn = v * s_inv[r];
      float w  = xn - BASEC * floorf(xn * (1.0f / BASEC));
      s_h1[r * 128 + wave * 16 + nn] = (bf16_t)(g * w + be);
    }
    __syncthreads();
    // GEMM2: this wave owns output columns [wave*128, wave*128+128) of D
#pragma unroll 1
    for (int t = 0; t < 8; ++t) {
      const int d0 = wave * 128 + t * 16;
      for (int kk = 0; kk < 4; ++kk) {
        v16bf a = frag16(s_h1, 128, nn, kk * 32, lane);
        v16bf b = frag16(w2p, FF, d0 + nn, ch * 128 + kk * 32, lane);
        c2[t] = wmma_bf16(a, b, c2[t]);
      }
    }
    __syncthreads();
  }

  // ================= epilogue: rot2 + mpn2 + tail =================
  if (tid < 16) s_sum[tid] = 0.0f;
  __syncthreads();
  float ps2[8];
#pragma unroll
  for (int j = 0; j < 8; ++j) ps2[j] = 0.0f;
#pragma unroll 1
  for (int t = 0; t < 8; ++t) {
    const int d = wave * 128 + t * 16 + nn;
    const float bias = b2[p * DD + d];
    const bool rot = (wave == 0) && (t == 0);      // columns 0..7 of D
    float cc = 1.0f, ss = 0.0f;
    if (rot && nn < 8) { float th = th2[p * KROT + (nn >> 1)]; cc = cosf(th); ss = sinf(th); }
#pragma unroll
    for (int j = 0; j < 8; ++j) {
      float v = c2[t][j] + bias;
      if (rot) {
        float part = __shfl_xor(v, 1, 32);
        if (nn < 8) v = (nn & 1) ? (cc * v + ss * part) : (cc * v - ss * part);
      }
      c2[t][j] = v;
      ps2[j] += v * v;
    }
  }
#pragma unroll
  for (int j = 0; j < 8; ++j) atomicAdd(&s_sum[rbase + j], ps2[j]);
  __syncthreads();
  if (tid < 16) s_inv[tid] = rsqrtf(s_sum[tid] * (1.0f / DD) + EPSC);
  __syncthreads();
#pragma unroll 1
  for (int t = 0; t < 8; ++t) {
    const int d  = wave * 128 + t * 16 + nn;
    const float g  = g2[p * DD + d];
    const float be = be2[p * DD + d];
    const float pb = pbias[p * DD + d];
    const float cx = ctx[p * DD + d];
#pragma unroll
    for (int j = 0; j < 8; ++j) {
      const int r = rbase + j;
      float xn = c2[t][j] * s_inv[r];
      float w  = xn - BASEC * floorf(xn * (1.0f / BASEC));
      float h  = g * w + be;
      h = h + pb;
      h = h + 0.1f * cx * tanhf(h);
      fbuf[((size_t)p * NROWS + (r0 + r)) * DD + d] = h;
    }
  }
}

// ---------------------------------------------------------------------------
// Stage 3: per-row mixture. One block per row (n = B*S).
// ---------------------------------------------------------------------------
__global__ __launch_bounds__(256)
void k_combine(const float* __restrict__ fbuf,
               const float* __restrict__ wpath, const float* __restrict__ bpath,
               const float* __restrict__ ovpath,
               const float* __restrict__ wteam, const float* __restrict__ bteam,
               const float* __restrict__ ovteam,
               float* __restrict__ out)
{
  const int row = blockIdx.x;
  const int tid = threadIdx.x;
  __shared__ float s_ps[PP];
  __shared__ float s_ts[3];
  if (tid < PP) s_ps[tid] = 0.0f;
  if (tid < 3)  s_ts[tid] = 0.0f;
  __syncthreads();

  float fv[PP][4];
  float acc[PP];
#pragma unroll
  for (int q = 0; q < PP; ++q) acc[q] = 0.0f;
#pragma unroll
  for (int i = 0; i < 4; ++i) {
    const int d = tid + i * 256;
    const float wp = wpath[d];
#pragma unroll
    for (int q = 0; q < PP; ++q) {
      float v = fbuf[((size_t)q * NROWS + row) * DD + d];
      fv[q][i] = v;
      acc[q] += v * wp;
    }
  }
#pragma unroll
  for (int q = 0; q < PP; ++q) atomicAdd(&s_ps[q], acc[q]);
  __syncthreads();

  // per-team softmax over path scores (computed redundantly per thread)
  float win[3][3];
#pragma unroll
  for (int t = 0; t < 3; ++t) {
    float e[3], m = -1e30f, sum = 0.0f;
#pragma unroll
    for (int s = 0; s < 3; ++s) {
      e[s] = s_ps[t * 3 + s] + bpath[0] - ovpath[t * 3 + s];
      m = fmaxf(m, e[s]);
    }
#pragma unroll
    for (int s = 0; s < 3; ++s) { e[s] = __expf(e[s] - m); sum += e[s]; }
#pragma unroll
    for (int s = 0; s < 3; ++s) win[t][s] = e[s] / sum;
  }

  float tf[3][4];
  float tacc[3] = {0.0f, 0.0f, 0.0f};
#pragma unroll
  for (int i = 0; i < 4; ++i) {
    const int d = tid + i * 256;
    const float wt = wteam[d];
#pragma unroll
    for (int t = 0; t < 3; ++t) {
      float v = fv[t * 3 + 0][i] * win[t][0] +
                fv[t * 3 + 1][i] * win[t][1] +
                fv[t * 3 + 2][i] * win[t][2];
      tf[t][i] = v;
      tacc[t] += v * wt;
    }
  }
#pragma unroll
  for (int t = 0; t < 3; ++t) atomicAdd(&s_ts[t], tacc[t]);
  __syncthreads();

  float wt3[3];
  {
    float e[3], m = -1e30f, sum = 0.0f;
#pragma unroll
    for (int t = 0; t < 3; ++t) {
      e[t] = s_ts[t] + bteam[0] - ovteam[t];
      m = fmaxf(m, e[t]);
    }
#pragma unroll
    for (int t = 0; t < 3; ++t) { e[t] = __expf(e[t] - m); sum += e[t]; }
#pragma unroll
    for (int t = 0; t < 3; ++t) wt3[t] = e[t] / sum;
  }
#pragma unroll
  for (int i = 0; i < 4; ++i) {
    const int d = tid + i * 256;
    out[(size_t)row * DD + d] =
        tf[0][i] * wt3[0] + tf[1][i] * wt3[1] + tf[2][i] * wt3[2];
  }
}

// ---------------------------------------------------------------------------
extern "C" void kernel_launch(void* const* d_in, const int* in_sizes, int n_in,
                              void* d_out, int out_size, void* d_ws, size_t ws_size,
                              hipStream_t stream) {
  (void)in_sizes; (void)n_in; (void)out_size; (void)ws_size;

  const float* x      = (const float*)d_in[0];
  const float* W1     = (const float*)d_in[1];
  const float* b1     = (const float*)d_in[2];
  const float* th1    = (const float*)d_in[3];
  const float* g1     = (const float*)d_in[4];
  const float* be1    = (const float*)d_in[5];
  const float* W2     = (const float*)d_in[6];
  const float* b2     = (const float*)d_in[7];
  const float* th2    = (const float*)d_in[8];
  const float* g2     = (const float*)d_in[9];
  const float* be2    = (const float*)d_in[10];
  const float* pbias  = (const float*)d_in[11];
  const float* ctx    = (const float*)d_in[12];
  const float* wpath  = (const float*)d_in[13];
  const float* bpath  = (const float*)d_in[14];
  const float* ovpath = (const float*)d_in[15];
  const float* wteam  = (const float*)d_in[16];
  const float* bteam  = (const float*)d_in[17];
  const float* ovteam = (const float*)d_in[18];

  // workspace layout (all offsets 256B aligned)
  char* ws = (char*)d_ws;
  const size_t xbytes = (size_t)NROWS * DD * sizeof(bf16_t);    // 16 MB
  const size_t wbytes = (size_t)PP * DD * FF * sizeof(bf16_t);  // 75.5 MB each
  bf16_t* xb  = (bf16_t*)ws;                           // [NROWS][D] bf16
  bf16_t* w1t = (bf16_t*)(ws + xbytes);                // [P][F][D] bf16
  bf16_t* w2t = (bf16_t*)(ws + xbytes + wbytes);       // [P][D][F] bf16
  float*  fbuf = (float*)(ws + xbytes + 2 * wbytes);   // [P][NROWS][D] f32

  // Stage 0/1: convert activations, convert+transpose weights (read once)
  k_convert_bf16<<<1024, 256, 0, stream>>>(x, xb, (size_t)NROWS * DD);
  k_transpose_to_bf16<<<2048, 256, 0, stream>>>(W1, w1t, DD, FF, PP);
  k_transpose_to_bf16<<<2048, 256, 0, stream>>>(W2, w2t, FF, DD, PP);

  // Stage 2: fused WMMA FFN, path-major block order for L2 weight residency
  k_path<<<dim3(PP * (NROWS / 16)), 256, 0, stream>>>(
      xb, w1t, w2t, b1, th1, g1, be1, b2, th2, g2, be2, pbias, ctx, fbuf);

  // Stage 3: mixture combine
  k_combine<<<NROWS, 256, 0, stream>>>(fbuf, wpath, bpath, ovpath,
                                       wteam, bteam, ovteam, (float*)d_out);
}
